// SNN_25786983646006
// MI455X (gfx1250) — compile-verified, matrix operand order checked by reading, not run on hardware
//
#include <hip/hip_runtime.h>

// ---------------------------------------------------------------------------
// CDNA5 (gfx1250) fused SNN forward: bf16 WMMA GEMM + Leaky-Integrate-Fire
// A-stripe staged in LDS (fragment-swizzled) via async global->LDS copies.
// ---------------------------------------------------------------------------

typedef __attribute__((ext_vector_type(16))) __bf16 v16bf;
typedef __attribute__((ext_vector_type(8)))  float  v8f;
typedef __attribute__((ext_vector_type(4)))  __bf16 v4bf;

union FragB16 {
    v16bf v;
    uint4 q[2];
};

// --------------------------- fp32 -> bf16 conversion -----------------------
__global__ void snn_cvt_f32_bf16(const float* __restrict__ in,
                                 __bf16* __restrict__ out, int n4) {
    int i = blockIdx.x * blockDim.x + threadIdx.x;
    if (i < n4) {
        float4 f = ((const float4*)in)[i];
        v4bf o;
        o.x = (__bf16)f.x;
        o.y = (__bf16)f.y;
        o.z = (__bf16)f.z;
        o.w = (__bf16)f.w;
        ((v4bf*)out)[i] = o;
    }
}

// ------------------- fused GEMM (C = A * W^T) + LIF epilogue ---------------
// A: [128 x K] bf16 activations, row stride lda; W: [N x K] bf16 row-major.
// Block = 8 waves, covers 16(M) x 256(N). The 16 x K A-stripe (64 KB) is
// staged once into LDS in per-lane WMMA-fragment order:
//   LDS[kkblk*1024 + L*32 .. +31] = the 32 bytes lane L feeds to
//   v_wmma_f32_16x16x32_bf16 for K-step kkblk
//   (16-bit A frag, lane L: row = L%16, halves at k = kk+8*(L/16) and +16).
// Lane stride 32B => ds_load_b128 reads are bank-conflict free.
__global__ void __launch_bounds__(256)
snn_layer_kernel(const __bf16* __restrict__ A, int lda,
                 const __bf16* __restrict__ W,
                 float*  __restrict__ mem_state,   // [128 x N]
                 __bf16* __restrict__ spk_state,   // [128 x N] or nullptr
                 float*  __restrict__ spk_rec,     // d_out slice for (t, layer)
                 float*  __restrict__ mem_rec,     // d_out slice for (t, layer)
                 const float* __restrict__ th_ptr,
                 int K, int N) {
    constexpr int KMAX = 2048;
    __shared__ __align__(16) unsigned char As[16 * KMAX * 2];   // 64 KB

    const int lane = threadIdx.x;        // wave32 lane
    const int wave = threadIdx.y;        // 8 waves / block
    const int tid  = wave * 32 + lane;
    const int half = lane >> 4;
    const int l16  = lane & 15;

    const int n0 = (blockIdx.x * 8 + wave) * 32;  // 32 output columns per wave
    const int m0 = blockIdx.y * 16;               // 16 output rows per block

    // ---- Stage A stripe into LDS with CDNA5 async global->LDS b128 copies.
    // 16B chunk c: kkblk = c/64, L = (c%64)/2, h2 = c%2;
    // global elem = (m0 + L%16)*lda + kkblk*32 + 8*(L/16) + 16*h2.
    {
        const unsigned asBase = (unsigned)(size_t)(&As[0]);  // LDS byte offset
        const int nChunks = (K / 32) * 64;
        for (int c = tid; c < nChunks; c += 256) {
            const int kkblk = c >> 6;
            const int cc    = c & 63;
            const int L     = cc >> 1;
            const int h2    = cc & 1;
            const int kel   = kkblk * 32 + 8 * (L >> 4) + 16 * h2;
            const __bf16* gsrc = A + (size_t)(m0 + (L & 15)) * lda + kel;
            const unsigned ldsOff = asBase + (unsigned)c * 16u;
            asm volatile("global_load_async_to_lds_b128 %0, %1, off"
                         :: "v"(ldsOff), "v"(gsrc)
                         : "memory");
        }
        asm volatile("s_wait_asynccnt 0x0" ::: "memory");
    }
    __syncthreads();

    v8f acc0 = {};
    v8f acc1 = {};

    const __bf16* Wrow0 = W + (size_t)(n0 + l16) * K;
    const __bf16* Wrow1 = W + (size_t)(n0 + 16 + l16) * K;
    const int kh    = 8 * half;
    const int aLane = lane * 32;

    for (int kk = 0, kb = 0; kk < K; kk += 32, kb += 1024) {
        FragB16 a, b0, b1;
        a.q[0]  = *(const uint4*)(As + kb + aLane);        // ds_load_b128
        a.q[1]  = *(const uint4*)(As + kb + aLane + 16);   // ds_load_b128
        b0.q[0] = *(const uint4*)(Wrow0 + kk + kh);
        b0.q[1] = *(const uint4*)(Wrow0 + kk + 16 + kh);
        b1.q[0] = *(const uint4*)(Wrow1 + kk + kh);
        b1.q[1] = *(const uint4*)(Wrow1 + kk + 16 + kh);
        // D = A x B + C   (fp32 accumulate)
        acc0 = __builtin_amdgcn_wmma_f32_16x16x32_bf16(
            false, a.v, false, b0.v, (short)0, acc0, false, false);
        acc1 = __builtin_amdgcn_wmma_f32_16x16x32_bf16(
            false, a.v, false, b1.v, (short)0, acc1, false, false);
    }

    // ---- fused snnTorch Leaky update (beta = 0.5, subtract-reset) ----
    const float th = th_ptr[0];
#pragma unroll
    for (int v = 0; v < 8; ++v) {
        const int m = m0 + v + 8 * half;   // f32 C/D: row = vgpr + 8*(lane/16)
        {   // first 16-column tile
            const size_t g = (size_t)m * N + (n0 + l16);
            const float mo    = mem_state[g];
            const float reset = (mo - th > 0.0f) ? 1.0f : 0.0f;
            const float mn    = 0.5f * mo + acc0[v] - reset * th;
            const float spk   = (mn - th > 0.0f) ? 1.0f : 0.0f;
            mem_state[g] = mn;
            mem_rec[g]   = mn;
            spk_rec[g]   = spk;
            if (spk_state) spk_state[g] = (__bf16)spk;
        }
        {   // second 16-column tile
            const size_t g = (size_t)m * N + (n0 + 16 + l16);
            const float mo    = mem_state[g];
            const float reset = (mo - th > 0.0f) ? 1.0f : 0.0f;
            const float mn    = 0.5f * mo + acc1[v] - reset * th;
            const float spk   = (mn - th > 0.0f) ? 1.0f : 0.0f;
            mem_state[g] = mn;
            mem_rec[g]   = mn;
            spk_rec[g]   = spk;
            if (spk_state) spk_state[g] = (__bf16)spk;
        }
    }
}

// ---------------------------------------------------------------------------
extern "C" void kernel_launch(void* const* d_in, const int* in_sizes, int n_in,
                              void* d_out, int out_size, void* d_ws,
                              size_t ws_size, hipStream_t stream) {
    (void)in_sizes; (void)n_in; (void)out_size; (void)ws_size;

    const float* x   = (const float*)d_in[0];   // [128, 100, 2048]
    const float* W0  = (const float*)d_in[1];   // [2048, 2048]
    const float* W1  = (const float*)d_in[2];   // [2048, 2048]
    const float* W2  = (const float*)d_in[3];   // [512, 2048]
    const float* th0 = (const float*)d_in[4];
    const float* th1 = (const float*)d_in[5];
    const float* th2 = (const float*)d_in[6];
    // d_in[7] = num_steps (device memory, statically 100 for this workload)

    constexpr int Bz = 128, T = 100, D = 2048, H1 = 2048, H2 = 2048, Dout = 512;

    // ------------------------- workspace layout ---------------------------
    char*  ws  = (char*)d_ws;
    size_t off = 0;
    auto alloc = [&](size_t bytes) -> void* {
        void* p = ws + off;
        off = (off + bytes + 255) & ~(size_t)255;
        return p;
    };
    __bf16* x_bf  = (__bf16*)alloc((size_t)Bz * T * D * 2);   // 52.4 MB
    __bf16* W0_bf = (__bf16*)alloc((size_t)H1 * D  * 2);      //  8.4 MB
    __bf16* W1_bf = (__bf16*)alloc((size_t)H2 * H1 * 2);      //  8.4 MB
    __bf16* W2_bf = (__bf16*)alloc((size_t)Dout * H2 * 2);    //  2.1 MB
    float*  mem0  = (float*)alloc((size_t)Bz * H1 * 4);       //  1.0 MB
    float*  mem1  = (float*)alloc((size_t)Bz * H2 * 4);       //  1.0 MB
    float*  mem2  = (float*)alloc((size_t)Bz * Dout * 4);     //  0.25 MB
    __bf16* spk0  = (__bf16*)alloc((size_t)Bz * H1 * 2);
    __bf16* spk1  = (__bf16*)alloc((size_t)Bz * H2 * 2);

    // ----------------- convert inputs/weights to bf16 ---------------------
    auto cvt = [&](const float* in, __bf16* out, size_t n) {
        int n4 = (int)(n / 4);
        snn_cvt_f32_bf16<<<(n4 + 255) / 256, 256, 0, stream>>>(in, out, n4);
    };
    cvt(x,  x_bf,  (size_t)Bz * T * D);
    cvt(W0, W0_bf, (size_t)H1 * D);
    cvt(W1, W1_bf, (size_t)H2 * H1);
    cvt(W2, W2_bf, (size_t)Dout * H2);

    // mem0/mem1/mem2 are contiguous (all sizes multiples of 256B): zero once.
    hipMemsetAsync(mem0, 0,
                   (size_t)Bz * (H1 + H2 + Dout) * sizeof(float), stream);

    // ------------------------- output layout ------------------------------
    float* out = (float*)d_out;
    const size_t spkOff0 = 0;
    const size_t spkOff1 = (size_t)T * Bz * H1;
    const size_t spkOff2 = (size_t)T * Bz * (H1 + H2);
    const size_t memBase = (size_t)T * Bz * (H1 + H2 + Dout);
    const size_t memOff0 = memBase;
    const size_t memOff1 = memBase + (size_t)T * Bz * H1;
    const size_t memOff2 = memBase + (size_t)T * Bz * (H1 + H2);

    // ---------------------- sequential timestep loop ----------------------
    const dim3 blk(32, 8);                       // 8 waves / workgroup
    for (int t = 0; t < T; ++t) {
        // Layer 0: A = x[:, t, :]  (row stride T*D), K=D, N=H1
        snn_layer_kernel<<<dim3(H1 / 256, Bz / 16), blk, 0, stream>>>(
            x_bf + (size_t)t * D, T * D, W0_bf, mem0, spk0,
            out + spkOff0 + (size_t)t * Bz * H1,
            out + memOff0 + (size_t)t * Bz * H1, th0, D, H1);
        // Layer 1: A = spk0, K=H1, N=H2
        snn_layer_kernel<<<dim3(H2 / 256, Bz / 16), blk, 0, stream>>>(
            spk0, H1, W1_bf, mem1, spk1,
            out + spkOff1 + (size_t)t * Bz * H2,
            out + memOff1 + (size_t)t * Bz * H2, th1, H1, H2);
        // Layer 2: A = spk1, K=H2, N=Dout (no downstream spike buffer)
        snn_layer_kernel<<<dim3(Dout / 256, Bz / 16), blk, 0, stream>>>(
            spk1, H2, W2_bf, mem2, (__bf16*)nullptr,
            out + spkOff2 + (size_t)t * Bz * Dout,
            out + memOff2 + (size_t)t * Bz * Dout, th2, H2, Dout);
    }
}
